// Attention_88046829568004
// MI455X (gfx1250) — matrix operand order repaired and checked by reading, not compiled, not run on hardware
//
#include <hip/hip_runtime.h>
#include <hip/hip_bf16.h>

// ---------- types ----------
typedef __attribute__((ext_vector_type(16))) __bf16        bf16x16;
typedef __attribute__((ext_vector_type(8)))  float         f32x8;
typedef __attribute__((ext_vector_type(4)))  float         f32x4;
typedef __attribute__((ext_vector_type(4)))  unsigned int  u32x4;
typedef __attribute__((ext_vector_type(2)))  unsigned int  u32x2;

union FragAB { bf16x16 v; u32x4 q[2]; unsigned short h[16]; };
union U128   { u32x4 q; unsigned short h[8]; };

// ---------- helpers ----------
__device__ __forceinline__ unsigned int f2b_u(float f) {
  unsigned int u = __float_as_uint(f);
  return (u + 0x7FFFu + ((u >> 16) & 1u)) >> 16;          // RNE f32 -> bf16
}
__device__ __forceinline__ unsigned short f2b(float f) { return (unsigned short)f2b_u(f); }
__device__ __forceinline__ unsigned int pack2(float a, float b) {
  return f2b_u(a) | (f2b_u(b) << 16);
}
// exchange with lane ^ 16 (SWAPX16: xor=0x10, and=0x1f -> offset 0x401F)
__device__ __forceinline__ float xor16(float x) {
  return __int_as_float(__builtin_amdgcn_ds_swizzle(__float_as_int(x), 0x401F));
}
__device__ __forceinline__ f32x8 wmma_bf16(const FragAB& a, const FragAB& b, f32x8 c) {
  return __builtin_amdgcn_wmma_f32_16x16x32_bf16(false, a.v, false, b.v, (short)0, c, false, false);
}

// ---------- constants ----------
#define HW   576          // 24*24 spatial tokens per image
#define CC   512          // channels
#define C3   1536         // 3*C
#define NB   8            // batch
#define NTOK 2304         // V_NUM*HW tokens per group
#define SB   (C3*HW)      // bf16 elements per batch in qkv ws

// =====================================================================
// bf16-WMMA GEMM:  Y[b] = A(f32,[M,512]) * B[b]   (per-batch B, K=512)
//   block = 8 waves as 4(M) x 2(N); each wave 32x32 (2x2 D-frags)
//   block tile 128(M) x 64(N); double-buffered LDS, 1 barrier / K-step
//   !B_BF16: B = x (f32 [512,HW]) , out bf16 ws      (QKV stage)
//    B_BF16: B = ao (bf16 [512,HW]), out f32 + bias  (proj stage)
// =====================================================================
template <bool B_BF16, bool PROJ>
__global__ __launch_bounds__(256) void gemm_k(
    const float* __restrict__ A, const float* __restrict__ Bf,
    const unsigned short* __restrict__ Bh, const float* __restrict__ bias,
    unsigned short* __restrict__ Obf, float* __restrict__ Of) {
  const int K = 512;
  const int t = threadIdx.x, lane = t & 31, w = t >> 5;
  const int wm = w >> 1, wn = w & 1;                 // 4 x 2 waves
  const int hi = (lane >> 4) & 1, lm = lane & 15;
  const int b  = blockIdx.z;
  const int m0 = blockIdx.x * 128;
  const int s0 = blockIdx.y * 64;

  __shared__ __align__(16) unsigned short As[2][128 * 32];  // [m][k] bf16
  __shared__ __align__(16) unsigned short Bs[2][64 * 32];   // [n][k] bf16 (transposed)

  const float* Ar = A + (size_t)m0 * K;
  const float* Bfp = B_BF16 ? nullptr : (Bf + (size_t)b * K * HW);
  const unsigned short* Bhp = B_BF16 ? (Bh + (size_t)b * K * HW) : nullptr;

  // staging registers (tile kk prefetched here before the LDS store)
  u32x2 ra[4];        // A: 4 reps x (4 f32 -> 2x packed bf16)
  f32x4 rbf[2];       // B f32 source: 2 reps x float4
  u32x4 rbh;          // B bf16 source: 8 bf16

  const int arow = t >> 3, ac = t & 7;               // A fill map (x4 reps of 32 rows)
  const int bkr  = B_BF16 ? (t >> 3) : (t >> 4);     // B fill map
  const int bc   = B_BF16 ? (t & 7) : (t & 15);

  // ---- preload K-step 0 ----
  #pragma unroll
  for (int rep = 0; rep < 4; ++rep) {
    f32x4 v = *(const f32x4*)(Ar + (size_t)(arow + rep * 32) * K + ac * 4);
    ra[rep].x = pack2(v.x, v.y); ra[rep].y = pack2(v.z, v.w);
  }
  if (!B_BF16) {
    #pragma unroll
    for (int rep = 0; rep < 2; ++rep) {
      int kr = bkr + rep * 16;
      rbf[rep] = *(const f32x4*)(Bfp + (size_t)kr * HW + s0 + bc * 4);
    }
  } else {
    rbh = *(const u32x4*)(Bhp + (size_t)bkr * HW + s0 + bc * 8);
  }

  f32x8 acc[2][2] = {};

  for (int kk = 0; kk < K; kk += 32) {
    const int sel = (kk >> 5) & 1;
    // ---- commit prefetched tile to LDS ----
    #pragma unroll
    for (int rep = 0; rep < 4; ++rep)
      *(u32x2*)&As[sel][(arow + rep * 32) * 32 + ac * 4] = ra[rep];
    if (!B_BF16) {
      #pragma unroll
      for (int rep = 0; rep < 2; ++rep) {
        int kr = bkr + rep * 16;
        Bs[sel][(bc * 4 + 0) * 32 + kr] = f2b(rbf[rep].x);
        Bs[sel][(bc * 4 + 1) * 32 + kr] = f2b(rbf[rep].y);
        Bs[sel][(bc * 4 + 2) * 32 + kr] = f2b(rbf[rep].z);
        Bs[sel][(bc * 4 + 3) * 32 + kr] = f2b(rbf[rep].w);
      }
    } else {
      U128 u; u.q = rbh;
      #pragma unroll
      for (int j = 0; j < 8; ++j) Bs[sel][(bc * 8 + j) * 32 + bkr] = u.h[j];
    }
    __syncthreads();

    // ---- prefetch next K-step while this one computes ----
    if (kk + 32 < K) {
      const int kn = kk + 32;
      #pragma unroll
      for (int rep = 0; rep < 4; ++rep) {
        f32x4 v = *(const f32x4*)(Ar + (size_t)(arow + rep * 32) * K + kn + ac * 4);
        ra[rep].x = pack2(v.x, v.y); ra[rep].y = pack2(v.z, v.w);
      }
      if (!B_BF16) {
        #pragma unroll
        for (int rep = 0; rep < 2; ++rep) {
          int kr = bkr + rep * 16;
          rbf[rep] = *(const f32x4*)(Bfp + (size_t)(kn + kr) * HW + s0 + bc * 4);
        }
      } else {
        rbh = *(const u32x4*)(Bhp + (size_t)(kn + bkr) * HW + s0 + bc * 8);
      }
    }

    // ---- 2x2 register-blocked WMMA ----
    FragAB af[2], bf[2];
    #pragma unroll
    for (int mi = 0; mi < 2; ++mi) {
      const int row = wm * 32 + mi * 16 + lm;
      af[mi].q[0] = *(const u32x4*)&As[sel][row * 32 + 8 * hi];
      af[mi].q[1] = *(const u32x4*)&As[sel][row * 32 + 16 + 8 * hi];
    }
    #pragma unroll
    for (int ni = 0; ni < 2; ++ni) {
      const int nrow = wn * 32 + ni * 16 + lm;
      bf[ni].q[0] = *(const u32x4*)&Bs[sel][nrow * 32 + 8 * hi];
      bf[ni].q[1] = *(const u32x4*)&Bs[sel][nrow * 32 + 16 + 8 * hi];
    }
    #pragma unroll
    for (int mi = 0; mi < 2; ++mi)
      #pragma unroll
      for (int ni = 0; ni < 2; ++ni)
        acc[mi][ni] = wmma_bf16(af[mi], bf[ni], acc[mi][ni]);
  }

  // ---- epilogue ----
  #pragma unroll
  for (int mi = 0; mi < 2; ++mi) {
    const int oc = m0 + wm * 32 + mi * 16;
    #pragma unroll
    for (int ni = 0; ni < 2; ++ni) {
      const int s = s0 + wn * 32 + ni * 16 + lm;
      if (!PROJ) {
        unsigned short* orow = Obf + (size_t)b * C3 * HW + (size_t)oc * HW + s;
        #pragma unroll
        for (int r = 0; r < 8; ++r)
          orow[(size_t)(r + 8 * hi) * HW] = f2b(acc[mi][ni][r]);
      } else {
        float* orow = Of + (size_t)b * CC * HW + (size_t)oc * HW + s;
        #pragma unroll
        for (int r = 0; r < 8; ++r) {
          int o = oc + r + 8 * hi;
          orow[(size_t)(r + 8 * hi) * HW] = acc[mi][ni][r] + bias[o];
        }
      }
    }
  }
}

// =====================================================================
// Flash attention per (g, head): S = scale*K^T Q, softmax over keys,
// O = V P.  Block = 128 queries (8 waves x 16 cols); key loop of 32,
// double-buffered K/V tiles (1 barrier per key block).
// qkv ws layout: bf16 [b][ch][s]; q ch = h*64+d, k +512, v +1024.
// =====================================================================
__global__ __launch_bounds__(256) void attn_k(
    const unsigned short* __restrict__ qkv, unsigned short* __restrict__ ao) {
  const int t = threadIdx.x, lane = t & 31, w = t >> 5;
  const int hi = (lane >> 4) & 1, lm = lane & 15;
  const int gh = blockIdx.y, g = gh >> 3, h = gh & 7;

  const int m  = blockIdx.x * 128 + w * 16 + lm;  // this lane's query column
  const int vbm = m / HW, sm = m % HW, bm = g * 4 + vbm;

  __shared__ __align__(16) unsigned short Ks[2][32 * 64];  // [n][d] bf16
  __shared__ __align__(16) unsigned short Vs[2][64 * 32];  // [d][n] bf16

  // ---- Q fragments (B-operand of S-gemm): K = d = dstep*32 + 8hi + (i<8 ? i : i+8) ----
  const unsigned short* qb = qkv + (size_t)bm * SB + (size_t)(h * 64) * HW + sm;
  FragAB qf[2];
  #pragma unroll
  for (int dstep = 0; dstep < 2; ++dstep)
    #pragma unroll
    for (int i = 0; i < 16; ++i)
      qf[dstep].h[i] = qb[(size_t)(dstep * 32 + hi * 8 + (i < 8 ? i : i + 8)) * HW];

  const int fd = t >> 2, fc = t & 3;      // cooperative fill map: 64 d x 32 n
  U128 rK; u32x4 rV;
  {   // preload key-block 0
    const unsigned short* kb = qkv + (size_t)(g * 4) * SB + (size_t)(512 + h * 64) * HW;
    const unsigned short* vb = qkv + (size_t)(g * 4) * SB + (size_t)(1024 + h * 64) * HW;
    rK.q = *(const u32x4*)(kb + (size_t)fd * HW + fc * 8);
    rV   = *(const u32x4*)(vb + (size_t)fd * HW + fc * 8);
  }

  f32x8 po[4] = {};            // O accum: 64(d) x 16(m) as 4 D-frags
  float mrun = -3.0e38f, lrun = 0.f;
  const float scale = 0.125f;  // hd^-0.5 = 1/8

  for (int nb = 0; nb < NTOK; nb += 32) {
    const int sel = (nb >> 5) & 1;
    // ---- commit prefetched K/V tile ----
    #pragma unroll
    for (int j = 0; j < 8; ++j) Ks[sel][(fc * 8 + j) * 64 + fd] = rK.h[j];  // transpose
    *(u32x4*)&Vs[sel][fd * 32 + fc * 8] = rV;
    __syncthreads();

    // ---- prefetch next key block ----
    if (nb + 32 < NTOK) {
      const int nn = nb + 32;
      const int vbn = nn / HW, sn = nn % HW, bn = g * 4 + vbn;
      const unsigned short* kb = qkv + (size_t)bn * SB + (size_t)(512 + h * 64) * HW + sn;
      const unsigned short* vb = qkv + (size_t)bn * SB + (size_t)(1024 + h * 64) * HW + sn;
      rK.q = *(const u32x4*)(kb + (size_t)fd * HW + fc * 8);
      rV   = *(const u32x4*)(vb + (size_t)fd * HW + fc * 8);
    }

    // ---- S tiles: two 16x16 n-subtiles, accumulated over d in 2 steps ----
    f32x8 s0v = {}, s1v = {};
    #pragma unroll
    for (int dstep = 0; dstep < 2; ++dstep) {
      FragAB a0, a1;
      a0.q[0] = *(const u32x4*)&Ks[sel][(lm) * 64 + dstep * 32 + 8 * hi];
      a0.q[1] = *(const u32x4*)&Ks[sel][(lm) * 64 + dstep * 32 + 16 + 8 * hi];
      a1.q[0] = *(const u32x4*)&Ks[sel][(16 + lm) * 64 + dstep * 32 + 8 * hi];
      a1.q[1] = *(const u32x4*)&Ks[sel][(16 + lm) * 64 + dstep * 32 + 16 + 8 * hi];
      s0v = wmma_bf16(a0, qf[dstep], s0v);
      s1v = wmma_bf16(a1, qf[dstep], s1v);
    }
    float s0[8], s1[8];
    #pragma unroll
    for (int r = 0; r < 8; ++r) { s0[r] = s0v[r] * scale; s1[r] = s1v[r] * scale; }

    // ---- online softmax over the 32 keys (column = m, rows split across lane pair) ----
    float mloc = -3.0e38f;
    #pragma unroll
    for (int r = 0; r < 8; ++r) mloc = fmaxf(mloc, fmaxf(s0[r], s1[r]));
    mloc = fmaxf(mloc, xor16(mloc));
    float mnew = fmaxf(mrun, mloc);
    float corr = __expf(mrun - mnew);
    float p0[8], p1[8], ps = 0.f;
    #pragma unroll
    for (int r = 0; r < 8; ++r) {
      p0[r] = __expf(s0[r] - mnew);
      p1[r] = __expf(s1[r] - mnew);
      ps += p0[r] + p1[r];
    }
    ps += xor16(ps);
    lrun = lrun * corr + ps;
    mrun = mnew;
    #pragma unroll
    for (int dc = 0; dc < 4; ++dc)
      #pragma unroll
      for (int r = 0; r < 8; ++r) po[dc][r] *= corr;

    // ---- rebuild P as B-fragment (K = key index 16hi+i) via lane-pair swap ----
    FragAB pf;
    #pragma unroll
    for (int r = 0; r < 8; ++r) {
      float o0 = xor16(p0[r]);
      float o1 = xor16(p1[r]);
      pf.h[r]     = f2b(hi ? o1 : p0[r]);  // K = 16hi + r
      pf.h[8 + r] = f2b(hi ? p1[r] : o0);  // K = 16hi + 8 + r
    }

    // ---- O += V * P : 4 d-chunks ----
    #pragma unroll
    for (int dc = 0; dc < 4; ++dc) {
      FragAB av;
      av.q[0] = *(const u32x4*)&Vs[sel][(dc * 16 + lm) * 32 + 16 * hi];
      av.q[1] = *(const u32x4*)&Vs[sel][(dc * 16 + lm) * 32 + 16 * hi + 8];
      po[dc] = wmma_bf16(av, pf, po[dc]);
    }
  }

  // ---- normalize and store bf16 [b][h*64+d][s] for the proj GEMM ----
  const float inv = 1.f / lrun;
  unsigned short* ab = ao + (size_t)bm * CC * HW + (size_t)(h * 64) * HW + sm;
  #pragma unroll
  for (int dc = 0; dc < 4; ++dc)
    #pragma unroll
    for (int r = 0; r < 8; ++r)
      ab[(size_t)(dc * 16 + r + 8 * hi) * HW] = f2b(po[dc][r] * inv);
}

// =====================================================================
extern "C" void kernel_launch(void* const* d_in, const int* in_sizes, int n_in,
                              void* d_out, int out_size, void* d_ws, size_t ws_size,
                              hipStream_t stream) {
  const float* x      = (const float*)d_in[0];   // [8,512,24,24]
  const float* w_qkv  = (const float*)d_in[1];   // [1536,512]
  const float* w_proj = (const float*)d_in[2];   // [512,512]
  const float* b_proj = (const float*)d_in[3];   // [512]
  float* out = (float*)d_out;                    // [8,512,24,24]

  unsigned short* qkv_ws = (unsigned short*)d_ws;                 // bf16 [8][1536][576]
  unsigned short* ao_ws  = qkv_ws + (size_t)NB * C3 * HW;         // bf16 [8][512][576]

  // Stage 1: QKV projection (per-batch GEMM 1536x512 * 512x576 -> bf16 ws)
  gemm_k<false, false><<<dim3(12, 9, NB), 256, 0, stream>>>(
      w_qkv, x, nullptr, nullptr, qkv_ws, nullptr);

  // Stage 2: flash attention per (group, head): 18 query-blocks x 16 (g,h)
  attn_k<<<dim3(18, 16, 1), 256, 0, stream>>>(qkv_ws, ao_ws);

  // Stage 3: output projection + bias (per-batch GEMM 512x512 * 512x576 -> f32)
  gemm_k<true, true><<<dim3(4, 9, NB), 256, 0, stream>>>(
      w_proj, nullptr, ao_ws, b_proj, nullptr, out);
}
